// LoRA_MOE_LM_764504179284
// MI455X (gfx1250) — matrix operand order repaired and checked by reading, not compiled, not run on hardware
//
#include <hip/hip_runtime.h>

// ---------------------------------------------------------------------------
// dims
// ---------------------------------------------------------------------------
#define NTOK 4096   // B*S
#define DMODEL 2048
#define MWID 5632
#define NEXP 8
#define RANK 16
#define KL 128      // E*R LoRA inner dim

typedef _Float16 h8  __attribute__((ext_vector_type(8)));
typedef _Float16 h16 __attribute__((ext_vector_type(16)));
typedef float    f8v __attribute__((ext_vector_type(8)));

// ---------------------------------------------------------------------------
// WMMA helpers (gfx1250: V_WMMA_F32_16X16X32_F16)
// ---------------------------------------------------------------------------
__device__ __forceinline__ f8v wmma16(h16 a, h16 b, f8v c) {
  return __builtin_amdgcn_wmma_f32_16x16x32_f16(false, a, false, b,
                                                (short)0, c, false, false);
}
// f16x2-split product: (ah+al)*(bh+bl) ~= ah*bh + ah*bl + al*bh
__device__ __forceinline__ f8v wmma3(h16 ah, h16 al, h16 bh, h16 bl, f8v c) {
  c = wmma16(ah, bh, c);
  c = wmma16(ah, bl, c);
  c = wmma16(al, bh, c);
  return c;
}

// LDS tile row stride (32 K halves + 16 pad, keeps 16B alignment, spreads banks)
#define LDT 48

// A fragment: 16x32 f16. lane<16: K 0..7 & 16..23 of row=lane&15;
// lane>=16: K 8..15 & 24..31. (ISA 7.12.2)
__device__ __forceinline__ h16 ld_afrag(const _Float16* s, int row, int lane) {
  int kb = (lane & 16) ? 8 : 0;
  h8 lo = *(const h8*)(s + row * LDT + kb);
  h8 hi = *(const h8*)(s + row * LDT + kb + 16);
  return __builtin_shufflevector(lo, hi, 0,1,2,3,4,5,6,7,8,9,10,11,12,13,14,15);
}
// B fragment: 32x16 f16. lane<16: K 0..7 & 16..23 of col=lane; lane>=16: K 8..15 & 24..31.
__device__ __forceinline__ h16 ld_bfrag(const _Float16* s, int col, int lane) {
  int kb = (lane & 16) ? 8 : 0;
  h8 lo = *(const h8*)(s + col * LDT + kb);
  h8 hi = *(const h8*)(s + col * LDT + kb + 16);
  return __builtin_shufflevector(lo, hi, 0,1,2,3,4,5,6,7,8,9,10,11,12,13,14,15);
}

// ---------------------------------------------------------------------------
// f32 -> (f16 hi, f16 lo residual) split
// ---------------------------------------------------------------------------
__global__ void k_split(const float* __restrict__ in, _Float16* __restrict__ hi,
                        _Float16* __restrict__ lo, int n) {
  for (int i = blockIdx.x * blockDim.x + threadIdx.x; i < n;
       i += gridDim.x * blockDim.x) {
    float f = in[i];
    _Float16 h = (_Float16)f;
    hi[i] = h;
    lo[i] = (_Float16)(f - (float)h);
  }
}

// Bexp [E, outdim, R] f32 -> Bflat [outdim, 128] f16  (col e*16+r)
__global__ void k_bflat(const float* __restrict__ Bexp, _Float16* __restrict__ out,
                        int outdim) {
  int n = outdim * KL;
  for (int i = blockIdx.x * blockDim.x + threadIdx.x; i < n;
       i += gridDim.x * blockDim.x) {
    int o = i >> 7, er = i & 127, e = er >> 4, r = er & 15;
    out[i] = (_Float16)Bexp[((size_t)e * outdim + o) * RANK + r];
  }
}

// ---------------------------------------------------------------------------
// Router: one wave per token. lane = (quarter q, expert e). Full f32.
// ---------------------------------------------------------------------------
__global__ __launch_bounds__(256) void k_router(const float* __restrict__ x,
                                                const float* __restrict__ Wr,
                                                const float* __restrict__ br,
                                                float* __restrict__ rout,
                                                float* __restrict__ choice) {
  int t = threadIdx.x, lane = t & 31, wv = t >> 5;
  int tok = blockIdx.x * 8 + wv;
  int e = lane & 7, q = lane >> 3;
  const float* xr = x + (size_t)tok * DMODEL + q * 512;
  const float* wr = Wr + (size_t)e * DMODEL + q * 512;
  float acc = 0.f;
  for (int i = 0; i < 512; i += 4) {
    float4 a = *(const float4*)(xr + i);
    float4 b = *(const float4*)(wr + i);
    acc += a.x * b.x + a.y * b.y + a.z * b.z + a.w * b.w;
  }
  acc += __shfl_xor(acc, 8);
  acc += __shfl_xor(acc, 16);               // lanes with same e now hold full dot
  float logit = acc + br[e];
  float mx = logit;
  mx = fmaxf(mx, __shfl_xor(mx, 1));
  mx = fmaxf(mx, __shfl_xor(mx, 2));
  mx = fmaxf(mx, __shfl_xor(mx, 4));
  float p = __expf(logit - mx);
  float s = p;
  s += __shfl_xor(s, 1); s += __shfl_xor(s, 2); s += __shfl_xor(s, 4);
  float rv = p / s;
  int flag = (logit == mx) ? e : 8;         // first-index argmax tie-break
  int o;
  o = __shfl_xor(flag, 1); flag = flag < o ? flag : o;
  o = __shfl_xor(flag, 2); flag = flag < o ? flag : o;
  o = __shfl_xor(flag, 4); flag = flag < o ? flag : o;
  if (q == 0) {
    rout[(size_t)tok * NEXP + e]   = rv;
    choice[(size_t)tok * NEXP + e] = (e == flag) ? 1.f : 0.f;
  }
}

// ---------------------------------------------------------------------------
// hg/hu = x @ {Ag,Au}^T (f32), then Z{g,u}[t, e*16+r] = 2*routing*h (f16).
// One wave per token; lane<16 -> gate rank r, lane>=16 -> up rank r.
// ---------------------------------------------------------------------------
__global__ __launch_bounds__(256) void k_hgu(const float* __restrict__ x,
                                             const float* __restrict__ Ag,
                                             const float* __restrict__ Au,
                                             const float* __restrict__ rout,
                                             _Float16* __restrict__ Zg,
                                             _Float16* __restrict__ Zu) {
  int t = threadIdx.x, lane = t & 31, wv = t >> 5;
  int tok = blockIdx.x * 8 + wv;
  int r = lane & 15, isU = lane >> 4;
  const float* Arow = (isU ? Au : Ag) + (size_t)r * DMODEL;
  const float* xr = x + (size_t)tok * DMODEL;
  float acc = 0.f;
  for (int i = 0; i < DMODEL; i += 4) {
    float4 a = *(const float4*)(xr + i);
    float4 w = *(const float4*)(Arow + i);
    acc += a.x * w.x + a.y * w.y + a.z * w.z + a.w * w.w;
  }
  _Float16* Z = isU ? Zu : Zg;
  const float* rr = rout + (size_t)tok * NEXP;
#pragma unroll
  for (int e = 0; e < NEXP; ++e)
    Z[(size_t)tok * KL + e * RANK + r] = (_Float16)(2.0f * rr[e] * acc);
}

// ---------------------------------------------------------------------------
// hd = h @ Ad^T from hi/lo f16 h; Zd built like above. Lanes split K in half.
// ---------------------------------------------------------------------------
__global__ __launch_bounds__(256) void k_hd(const _Float16* __restrict__ hh,
                                            const _Float16* __restrict__ hl,
                                            const float* __restrict__ Ad,
                                            const float* __restrict__ rout,
                                            _Float16* __restrict__ Zd) {
  int t = threadIdx.x, lane = t & 31, wv = t >> 5;
  int tok = blockIdx.x * 8 + wv;
  int r = lane & 15, half = lane >> 4;
  const _Float16* ph = hh + (size_t)tok * MWID + half * (MWID / 2);
  const _Float16* pl = hl + (size_t)tok * MWID + half * (MWID / 2);
  const float* Arow = Ad + (size_t)r * MWID + half * (MWID / 2);
  float acc = 0.f;
  for (int i = 0; i < MWID / 2; i += 8) {
    h8 a = *(const h8*)(ph + i);
    h8 b = *(const h8*)(pl + i);
#pragma unroll
    for (int k = 0; k < 8; ++k)
      acc += ((float)a[k] + (float)b[k]) * Arow[i + k];
  }
  acc += __shfl_xor(acc, 16);
  if (half == 0) {
    const float* rr = rout + (size_t)tok * NEXP;
#pragma unroll
    for (int e = 0; e < NEXP; ++e)
      Zd[(size_t)tok * KL + e * RANK + r] = (_Float16)(2.0f * rr[e] * acc);
  }
}

// ---------------------------------------------------------------------------
// Fused gate & up GEMM + SwiGLU.  Block tile: 128 tok x 64 cols (both mats).
// 8 waves, each owns 2x2 16x16 tiles per matrix. f16x2 split -> 3 WMMA/tile.
// ---------------------------------------------------------------------------
__global__ __launch_bounds__(256, 1) void k_gateup(
    const _Float16* __restrict__ xh, const _Float16* __restrict__ xl,
    const _Float16* __restrict__ Wgh, const _Float16* __restrict__ Wgl,
    const _Float16* __restrict__ Wuh, const _Float16* __restrict__ Wul,
    const _Float16* __restrict__ Zg, const _Float16* __restrict__ Zu,
    const _Float16* __restrict__ Bgf, const _Float16* __restrict__ Buf,
    _Float16* __restrict__ hh, _Float16* __restrict__ hl) {
  __shared__ _Float16 sAh[128 * LDT], sAl[128 * LDT];
  __shared__ _Float16 sBh[2][64 * LDT], sBl[2][64 * LDT];  // [0]=gate [1]=up

  const int t = threadIdx.x, lane = t & 31, wv = t >> 5;
  const int tok0 = blockIdx.y * 128;
  const int col0 = blockIdx.x * 64;
  const int rb0 = (wv & 3) * 2, cb0 = (wv >> 2) * 2;
  const int m15 = lane & 15;
  const int r0 = rb0 * 16 + m15, r1 = r0 + 16;
  const int c0 = cb0 * 16 + m15, c1 = c0 + 16;

  f8v accG[2][2] = {};
  f8v accU[2][2] = {};

  for (int k0 = 0; k0 < DMODEL; k0 += 32) {
    __syncthreads();
#pragma unroll
    for (int jj = 0; jj < 8; ++jj) {              // 2048 half8 chunks
      int idx = t + jj * 256;
      const _Float16* src; _Float16* dst;
      if (idx < 512)       { int l = idx;        int row = l >> 2, c = l & 3; src = xh  + (size_t)(tok0 + row) * DMODEL + k0 + c * 8; dst = sAh    + row * LDT + c * 8; }
      else if (idx < 1024) { int l = idx - 512;  int row = l >> 2, c = l & 3; src = xl  + (size_t)(tok0 + row) * DMODEL + k0 + c * 8; dst = sAl    + row * LDT + c * 8; }
      else if (idx < 1280) { int l = idx - 1024; int row = l >> 2, c = l & 3; src = Wgh + (size_t)(col0 + row) * DMODEL + k0 + c * 8; dst = sBh[0] + row * LDT + c * 8; }
      else if (idx < 1536) { int l = idx - 1280; int row = l >> 2, c = l & 3; src = Wgl + (size_t)(col0 + row) * DMODEL + k0 + c * 8; dst = sBl[0] + row * LDT + c * 8; }
      else if (idx < 1792) { int l = idx - 1536; int row = l >> 2, c = l & 3; src = Wuh + (size_t)(col0 + row) * DMODEL + k0 + c * 8; dst = sBh[1] + row * LDT + c * 8; }
      else                 { int l = idx - 1792; int row = l >> 2, c = l & 3; src = Wul + (size_t)(col0 + row) * DMODEL + k0 + c * 8; dst = sBl[1] + row * LDT + c * 8; }
      *(h8*)dst = *(const h8*)src;
    }
    if (k0 + 32 < DMODEL)                         // gfx1250 global_prefetch
      __builtin_prefetch(xh + (size_t)(tok0 + (t >> 1)) * DMODEL + k0 + 32, 0, 0);
    __syncthreads();

    h16 a0h = ld_afrag(sAh, r0, lane), a1h = ld_afrag(sAh, r1, lane);
    h16 a0l = ld_afrag(sAl, r0, lane), a1l = ld_afrag(sAl, r1, lane);
    h16 g0h = ld_bfrag(sBh[0], c0, lane), g1h = ld_bfrag(sBh[0], c1, lane);
    h16 g0l = ld_bfrag(sBl[0], c0, lane), g1l = ld_bfrag(sBl[0], c1, lane);
    h16 u0h = ld_bfrag(sBh[1], c0, lane), u1h = ld_bfrag(sBh[1], c1, lane);
    h16 u0l = ld_bfrag(sBl[1], c0, lane), u1l = ld_bfrag(sBl[1], c1, lane);

    accG[0][0] = wmma3(a0h, a0l, g0h, g0l, accG[0][0]);
    accG[0][1] = wmma3(a0h, a0l, g1h, g1l, accG[0][1]);
    accG[1][0] = wmma3(a1h, a1l, g0h, g0l, accG[1][0]);
    accG[1][1] = wmma3(a1h, a1l, g1h, g1l, accG[1][1]);
    accU[0][0] = wmma3(a0h, a0l, u0h, u0l, accU[0][0]);
    accU[0][1] = wmma3(a0h, a0l, u1h, u1l, accU[0][1]);
    accU[1][0] = wmma3(a1h, a1l, u0h, u0l, accU[1][0]);
    accU[1][1] = wmma3(a1h, a1l, u1h, u1l, accU[1][1]);
  }

  // LoRA tails (K=128, single-f16 precision is ample for the small term)
#pragma unroll 1
  for (int mat = 0; mat < 2; ++mat) {
    const _Float16* Zs = mat ? Zu : Zg;
    const _Float16* Bs = mat ? Buf : Bgf;
    for (int k0 = 0; k0 < KL; k0 += 32) {
      __syncthreads();
#pragma unroll
      for (int jj = 0; jj < 3; ++jj) {            // 768 half8 chunks
        int idx = t + jj * 256;
        if (idx < 512) { int row = idx >> 2, c = idx & 3; *(h8*)(sAh + row * LDT + c * 8) = *(const h8*)(Zs + (size_t)(tok0 + row) * KL + k0 + c * 8); }
        else { int l = idx - 512; int row = l >> 2, c = l & 3; *(h8*)(sBh[0] + row * LDT + c * 8) = *(const h8*)(Bs + (size_t)(col0 + row) * KL + k0 + c * 8); }
      }
      __syncthreads();
      h16 a0 = ld_afrag(sAh, r0, lane), a1 = ld_afrag(sAh, r1, lane);
      h16 b0 = ld_bfrag(sBh[0], c0, lane), b1 = ld_bfrag(sBh[0], c1, lane);
      f8v (*acc)[2] = mat ? accU : accG;
      acc[0][0] = wmma16(a0, b0, acc[0][0]);
      acc[0][1] = wmma16(a0, b1, acc[0][1]);
      acc[1][0] = wmma16(a1, b0, acc[1][0]);
      acc[1][1] = wmma16(a1, b1, acc[1][1]);
    }
  }

  // epilogue: h = silu(gate)*up, stored as f16 hi/lo (A operand of down proj)
  const int rowoff = (lane & 16) ? 8 : 0;
#pragma unroll
  for (int i = 0; i < 2; ++i)
#pragma unroll
    for (int j = 0; j < 2; ++j)
#pragma unroll
      for (int r = 0; r < 8; ++r) {
        float g = accG[i][j][r], u = accU[i][j][r];
        float hv = (g / (1.f + __expf(-g))) * u;
        _Float16 hhi = (_Float16)hv;
        _Float16 hlo = (_Float16)(hv - (float)hhi);
        size_t o = (size_t)(tok0 + (rb0 + i) * 16 + rowoff + r) * MWID +
                   (col0 + (cb0 + j) * 16 + m15);
        hh[o] = hhi;
        hl[o] = hlo;
      }
}

// ---------------------------------------------------------------------------
// Down projection: out = h @ Wd^T + Zd @ Bd^T. Same tiling, one matrix.
// ---------------------------------------------------------------------------
__global__ __launch_bounds__(256, 1) void k_down(
    const _Float16* __restrict__ hh, const _Float16* __restrict__ hl,
    const _Float16* __restrict__ Wdh, const _Float16* __restrict__ Wdl,
    const _Float16* __restrict__ Zd, const _Float16* __restrict__ Bdf,
    float* __restrict__ out) {
  __shared__ _Float16 sAh[128 * LDT], sAl[128 * LDT];
  __shared__ _Float16 sBh[64 * LDT], sBl[64 * LDT];

  const int t = threadIdx.x, lane = t & 31, wv = t >> 5;
  const int tok0 = blockIdx.y * 128;
  const int col0 = blockIdx.x * 64;
  const int rb0 = (wv & 3) * 2, cb0 = (wv >> 2) * 2;
  const int m15 = lane & 15;
  const int r0 = rb0 * 16 + m15, r1 = r0 + 16;
  const int c0 = cb0 * 16 + m15, c1 = c0 + 16;

  f8v acc[2][2] = {};

  for (int k0 = 0; k0 < MWID; k0 += 32) {
    __syncthreads();
#pragma unroll
    for (int jj = 0; jj < 6; ++jj) {              // 1536 half8 chunks
      int idx = t + jj * 256;
      const _Float16* src; _Float16* dst;
      if (idx < 512)       { int l = idx;        int row = l >> 2, c = l & 3; src = hh  + (size_t)(tok0 + row) * MWID + k0 + c * 8; dst = sAh + row * LDT + c * 8; }
      else if (idx < 1024) { int l = idx - 512;  int row = l >> 2, c = l & 3; src = hl  + (size_t)(tok0 + row) * MWID + k0 + c * 8; dst = sAl + row * LDT + c * 8; }
      else if (idx < 1280) { int l = idx - 1024; int row = l >> 2, c = l & 3; src = Wdh + (size_t)(col0 + row) * MWID + k0 + c * 8; dst = sBh + row * LDT + c * 8; }
      else                 { int l = idx - 1280; int row = l >> 2, c = l & 3; src = Wdl + (size_t)(col0 + row) * MWID + k0 + c * 8; dst = sBl + row * LDT + c * 8; }
      *(h8*)dst = *(const h8*)src;
    }
    if (k0 + 32 < MWID)
      __builtin_prefetch(hh + (size_t)(tok0 + (t >> 1)) * MWID + k0 + 32, 0, 0);
    __syncthreads();

    h16 a0h = ld_afrag(sAh, r0, lane), a1h = ld_afrag(sAh, r1, lane);
    h16 a0l = ld_afrag(sAl, r0, lane), a1l = ld_afrag(sAl, r1, lane);
    h16 b0h = ld_bfrag(sBh, c0, lane), b1h = ld_bfrag(sBh, c1, lane);
    h16 b0l = ld_bfrag(sBl, c0, lane), b1l = ld_bfrag(sBl, c1, lane);

    acc[0][0] = wmma3(a0h, a0l, b0h, b0l, acc[0][0]);
    acc[0][1] = wmma3(a0h, a0l, b1h, b1l, acc[0][1]);
    acc[1][0] = wmma3(a1h, a1l, b0h, b0l, acc[1][0]);
    acc[1][1] = wmma3(a1h, a1l, b1h, b1l, acc[1][1]);
  }

  // LoRA tail K=128
  for (int k0 = 0; k0 < KL; k0 += 32) {
    __syncthreads();
#pragma unroll
    for (int jj = 0; jj < 3; ++jj) {
      int idx = t + jj * 256;
      if (idx < 512) { int row = idx >> 2, c = idx & 3; *(h8*)(sAh + row * LDT + c * 8) = *(const h8*)(Zd + (size_t)(tok0 + row) * KL + k0 + c * 8); }
      else { int l = idx - 512; int row = l >> 2, c = l & 3; *(h8*)(sBh + row * LDT + c * 8) = *(const h8*)(Bdf + (size_t)(col0 + row) * KL + k0 + c * 8); }
    }
    __syncthreads();
    h16 a0 = ld_afrag(sAh, r0, lane), a1 = ld_afrag(sAh, r1, lane);
    h16 b0 = ld_bfrag(sBh, c0, lane), b1 = ld_bfrag(sBh, c1, lane);
    acc[0][0] = wmma16(a0, b0, acc[0][0]);
    acc[0][1] = wmma16(a0, b1, acc[0][1]);
    acc[1][0] = wmma16(a1, b0, acc[1][0]);
    acc[1][1] = wmma16(a1, b1, acc[1][1]);
  }

  const int rowoff = (lane & 16) ? 8 : 0;
#pragma unroll
  for (int i = 0; i < 2; ++i)
#pragma unroll
    for (int j = 0; j < 2; ++j)
#pragma unroll
      for (int r = 0; r < 8; ++r)
        out[(size_t)(tok0 + (rb0 + i) * 16 + rowoff + r) * DMODEL +
            (col0 + (cb0 + j) * 16 + m15)] = acc[i][j][r];
}

// ---------------------------------------------------------------------------
// launch
// ---------------------------------------------------------------------------
extern "C" void kernel_launch(void* const* d_in, const int* in_sizes, int n_in,
                              void* d_out, int out_size, void* d_ws, size_t ws_size,
                              hipStream_t stream) {
  const float* x  = (const float*)d_in[0];
  const float* Wr = (const float*)d_in[1];
  const float* br = (const float*)d_in[2];
  const float* Wg = (const float*)d_in[3];
  const float* Wu = (const float*)d_in[4];
  const float* Wd = (const float*)d_in[5];
  const float* Ag = (const float*)d_in[6];
  const float* Au = (const float*)d_in[7];
  const float* Ad = (const float*)d_in[8];
  const float* Bg = (const float*)d_in[9];
  const float* Bu = (const float*)d_in[10];
  const float* Bd = (const float*)d_in[11];

  float* out    = (float*)d_out;                      // [4096, 2048]
  float* rout   = out + (size_t)NTOK * DMODEL;        // [4096, 8]
  float* choice = rout + (size_t)NTOK * NEXP;         // [4096, 8]

  char* w = (char*)d_ws;
  size_t off = 0;
  auto alloc = [&](size_t bytes) -> void* {
    void* p = w + off;
    off += (bytes + 255) & ~(size_t)255;
    return p;
  };
  _Float16* xh  = (_Float16*)alloc((size_t)NTOK * DMODEL * 2);
  _Float16* xl  = (_Float16*)alloc((size_t)NTOK * DMODEL * 2);
  _Float16* Wgh = (_Float16*)alloc((size_t)MWID * DMODEL * 2);
  _Float16* Wgl = (_Float16*)alloc((size_t)MWID * DMODEL * 2);
  _Float16* Wuh = (_Float16*)alloc((size_t)MWID * DMODEL * 2);
  _Float16* Wul = (_Float16*)alloc((size_t)MWID * DMODEL * 2);
  _Float16* Wdh = (_Float16*)alloc((size_t)DMODEL * MWID * 2);
  _Float16* Wdl = (_Float16*)alloc((size_t)DMODEL * MWID * 2);
  _Float16* Bgf = (_Float16*)alloc((size_t)MWID * KL * 2);
  _Float16* Buf = (_Float16*)alloc((size_t)MWID * KL * 2);
  _Float16* Bdf = (_Float16*)alloc((size_t)DMODEL * KL * 2);
  _Float16* Zg  = (_Float16*)alloc((size_t)NTOK * KL * 2);
  _Float16* Zu  = (_Float16*)alloc((size_t)NTOK * KL * 2);
  _Float16* Zd  = (_Float16*)alloc((size_t)NTOK * KL * 2);
  _Float16* hh  = (_Float16*)alloc((size_t)NTOK * MWID * 2);
  _Float16* hl  = (_Float16*)alloc((size_t)NTOK * MWID * 2);

  k_split<<<4096, 256, 0, stream>>>(x,  xh,  xl,  NTOK * DMODEL);
  k_split<<<4096, 256, 0, stream>>>(Wg, Wgh, Wgl, MWID * DMODEL);
  k_split<<<4096, 256, 0, stream>>>(Wu, Wuh, Wul, MWID * DMODEL);
  k_split<<<4096, 256, 0, stream>>>(Wd, Wdh, Wdl, DMODEL * MWID);
  k_bflat<<<1024, 256, 0, stream>>>(Bg, Bgf, MWID);
  k_bflat<<<1024, 256, 0, stream>>>(Bu, Buf, MWID);
  k_bflat<<<512,  256, 0, stream>>>(Bd, Bdf, DMODEL);

  k_router<<<NTOK / 8, 256, 0, stream>>>(x, Wr, br, rout, choice);
  k_hgu<<<NTOK / 8, 256, 0, stream>>>(x, Ag, Au, rout, Zg, Zu);

  k_gateup<<<dim3(MWID / 64, NTOK / 128), 256, 0, stream>>>(
      xh, xl, Wgh, Wgl, Wuh, Wul, Zg, Zu, Bgf, Buf, hh, hl);

  k_hd<<<NTOK / 8, 256, 0, stream>>>(hh, hl, Ad, rout, Zd);

  k_down<<<dim3(DMODEL / 64, NTOK / 128), 256, 0, stream>>>(
      hh, hl, Wdh, Wdl, Zd, Bdf, out);
}
